// SparseAttention_86406152061631
// MI455X (gfx1250) — matrix operand order repaired and checked by reading, not compiled, not run on hardware
//
#include <hip/hip_runtime.h>

typedef _Float16 half_t;
typedef __attribute__((ext_vector_type(16))) _Float16 v16h;
typedef __attribute__((ext_vector_type(8)))  _Float16 v8h;
typedef __attribute__((ext_vector_type(4)))  _Float16 v4h;
typedef __attribute__((ext_vector_type(4)))  float    v4f;
typedef __attribute__((ext_vector_type(8)))  float    v8f;

#define S_LEN   2048
#define HID     1024
#define NH      16
#define HD      64
#define WIN     128
#define GSTRIDE 64
#define NGLOB   (S_LEN / GSTRIDE)   // 32 global tokens
#define NEG_INF (-1.0e9f)

// ---------------------------------------------------------------- WMMA helpers

__device__ __forceinline__ v8f wmma_f16(v16h a, v16h b, v8f c) {
  // D = A(16x32 f16) * B(32x16 f16) + C(16x16 f32)
  return __builtin_amdgcn_wmma_f32_16x16x32_f16(
      /*neg_a=*/false, a, /*neg_b=*/false, b,
      /*c_mod=*/(short)0, c, /*reuse_a=*/false, /*reuse_b=*/false);
}

// A fragment: 16x32 f16, row-major source, row stride ld (elements).
// ISA layout: lane<16 -> row M=lane, K {0..7,16..23}; lane>=16 -> K {8..15,24..31}
__device__ __forceinline__ v16h load_a16x32(const half_t* __restrict__ base, int ld) {
  const int lane = threadIdx.x & 31;
  const int m  = lane & 15;
  const int ko = (lane >> 4) * 8;
  v8h lo = *(const v8h*)(base + (size_t)m * ld + ko);
  v8h hi = *(const v8h*)(base + (size_t)m * ld + 16 + ko);
  v16h r;
#pragma unroll
  for (int i = 0; i < 8; ++i) { r[i] = lo[i]; r[8 + i] = hi[i]; }
  return r;
}

// B fragment: 32x16 f16 where column n of B is 16(x2) contiguous elements at
// base + n*ld  (i.e. B = W^T with W row-major, or V from a d-major buffer).
// ISA layout: lane<16 -> col N=lane, K {0..7,16..23}; lane>=16 -> col N=lane-16, K {8..15,24..31}
__device__ __forceinline__ v16h load_b32x16(const half_t* __restrict__ base, int ld) {
  const int lane = threadIdx.x & 31;
  const int n  = lane & 15;
  const int kb = (lane >> 4) * 8;
  v8h lo = *(const v8h*)(base + (size_t)n * ld + kb);
  v8h hi = *(const v8h*)(base + (size_t)n * ld + 16 + kb);
  v16h r;
#pragma unroll
  for (int i = 0; i < 8; ++i) { r[i] = lo[i]; r[8 + i] = hi[i]; }
  return r;
}

// ---- 32x64 tile GEMM, 2-stage pipeline with sched barriers ----------------
// Loads for the next K-chunk are issued ahead of the current chunk's WMMAs;
// sched_barrier(0) after each compute pins the in-place reload AFTER the
// compute that kills those registers (no temp copies, no WMMA->VALU NOPs),
// keeping VGPR pressure under 256.

struct Frags { v16h a0, a1, b0, b1, b2, b3; };

__device__ __forceinline__ Frags load_set(const half_t* __restrict__ A,
                                          const half_t* __restrict__ W, int kk) {
  Frags f;
  f.a0 = load_a16x32(A + kk, HID);
  f.a1 = load_a16x32(A + (size_t)16 * HID + kk, HID);
  f.b0 = load_b32x16(W + kk, HID);
  f.b1 = load_b32x16(W + (size_t)16 * HID + kk, HID);
  f.b2 = load_b32x16(W + (size_t)32 * HID + kk, HID);
  f.b3 = load_b32x16(W + (size_t)48 * HID + kk, HID);
  return f;
}

__device__ __forceinline__ void compute_set(const Frags& f, v8f acc[2][4]) {
  acc[0][0] = wmma_f16(f.a0, f.b0, acc[0][0]);
  acc[1][0] = wmma_f16(f.a1, f.b0, acc[1][0]);
  acc[0][1] = wmma_f16(f.a0, f.b1, acc[0][1]);
  acc[1][1] = wmma_f16(f.a1, f.b1, acc[1][1]);
  acc[0][2] = wmma_f16(f.a0, f.b2, acc[0][2]);
  acc[1][2] = wmma_f16(f.a1, f.b2, acc[1][2]);
  acc[0][3] = wmma_f16(f.a0, f.b3, acc[0][3]);
  acc[1][3] = wmma_f16(f.a1, f.b3, acc[1][3]);
}

__device__ __forceinline__ void gemm_32x64_acc(
    const half_t* __restrict__ A, const half_t* __restrict__ W, v8f acc[2][4]) {
  Frags cur = load_set(A, W, 0);
#pragma unroll 1
  for (int kk = 0; kk + 64 < HID; kk += 64) {
    Frags nxt = load_set(A, W, kk + 32);   // loads fly over compute(cur)
    compute_set(cur, acc);
    __builtin_amdgcn_sched_barrier(0);     // keep cur-reload below compute(cur)
    cur = load_set(A, W, kk + 64);         // loads fly over compute(nxt)
    compute_set(nxt, acc);
    __builtin_amdgcn_sched_barrier(0);
  }
  Frags last = load_set(A, W, HID - 32);
  compute_set(cur, acc);
  compute_set(last, acc);
}

// ---------------------------------------------------------------- kernels

__global__ __launch_bounds__(256) void cvt_f32_f16(const float* __restrict__ src,
                                                   half_t* __restrict__ dst, int n) {
  int i = (blockIdx.x * blockDim.x + threadIdx.x) * 4;
  if (i >= n) return;
  v4f v = *(const v4f*)(src + i);
  *(v4h*)(dst + i) = __builtin_convertvector(v, v4h);
}

// qkv = x @ Wqkv^T ; scatter into Q[h][s][d], K[h][s][d], Vt[h][d][s] (f16)
__global__ __launch_bounds__(128) void qkv_gemm(
    const half_t* __restrict__ xh, const half_t* __restrict__ wh,
    half_t* __restrict__ Qh, half_t* __restrict__ Kh, half_t* __restrict__ Vt) {
  const int wave = blockIdx.x * (blockDim.x >> 5) + (threadIdx.x >> 5);
  const int NT = (3 * HID) / 64;              // 48 col-tiles of 64
  const int mt = wave / NT, nt = wave % NT;
  const int s0 = mt * 32, o0 = nt * 64;
  const int lane = threadIdx.x & 31;
  const int n  = lane & 15;
  const int rb = (lane >> 4) * 8;

  v8f acc[2][4] = {};
  gemm_32x64_acc(xh + (size_t)s0 * HID, wh + (size_t)o0 * HID, acc);

  const int sec = o0 / HID;                   // 0=Q 1=K 2=V (64-wide tile = 1 head)
  const int h   = (o0 % HID) / HD;
  if (sec < 2) {
    half_t* dst = (sec == 0 ? Qh : Kh) + (size_t)h * S_LEN * HD;
#pragma unroll
    for (int i = 0; i < 2; ++i)
#pragma unroll
      for (int t = 0; t < 4; ++t)
#pragma unroll
        for (int r = 0; r < 8; ++r)
          dst[(size_t)(s0 + i * 16 + rb + r) * HD + t * 16 + n] =
              (half_t)acc[i][t][r];
  } else {                                    // V stored d-major for PV B-frags
    half_t* dst = Vt + (size_t)h * HD * S_LEN;
#pragma unroll
    for (int i = 0; i < 2; ++i)
#pragma unroll
      for (int t = 0; t < 4; ++t) {
        v8h pk;
#pragma unroll
        for (int r = 0; r < 8; ++r) pk[r] = (half_t)acc[i][t][r];
        *(v8h*)(dst + (size_t)(t * 16 + n) * S_LEN + s0 + i * 16 + rb) = pk;
      }
  }
}

// Compact copies of the 32 global tokens: Kg[h][g][d], Vgt[h][d][g]
__global__ __launch_bounds__(256) void gather_glb(
    const half_t* __restrict__ Kh, const half_t* __restrict__ Vt,
    half_t* __restrict__ Kg, half_t* __restrict__ Vgt) {
  int idx = blockIdx.x * blockDim.x + threadIdx.x;
  if (idx >= NH * NGLOB * HD) return;
  int h = idx / (NGLOB * HD);
  int rem = idx % (NGLOB * HD);
  int g = rem / HD, d = rem % HD;
  Kg[((size_t)h * NGLOB + g) * HD + d] =
      Kh[((size_t)h * S_LEN + g * GSTRIDE) * HD + d];
  Vgt[((size_t)h * HD + d) * NGLOB + g] =
      Vt[((size_t)h * HD + d) * S_LEN + g * GSTRIDE];
}

// Flash-style masked attention, one wave per (head, 16-query tile).
__global__ __launch_bounds__(32) void attn_sparse(
    const half_t* __restrict__ Qh, const half_t* __restrict__ Kh,
    const half_t* __restrict__ Vt, const half_t* __restrict__ Kg,
    const half_t* __restrict__ Vgt, half_t* __restrict__ attn_h) {
  __shared__ float  sc[16][32];   // raw scores (f32)
  __shared__ half_t pf[16][32];   // softmax numerators (f16)
  __shared__ float  alf[16];      // per-row rescale
  __shared__ float  linv[16];     // final 1/l

  const int wid = blockIdx.x;
  const int h   = wid >> 7;
  const int q0  = (wid & 127) * 16;
  const int lane = threadIdx.x;
  const int n  = lane & 15;
  const int rb = (lane >> 4) * 8;

  const half_t* Qb = Qh + ((size_t)h * S_LEN + q0) * HD;
  const v16h aq0 = load_a16x32(Qb, HD);       // d 0..31
  const v16h aq1 = load_a16x32(Qb + 32, HD);  // d 32..63

  v8f acc[4] = {};
  float m_i = -3.0e38f, l_i = 0.0f;           // per-row state (lanes 0..15)

  auto process = [&](const half_t* kT, const half_t* vT, int vld,
                     int jbase, int jstep, int mode, int klo, int khi) {
    // Issue ALL loads up front: the 4 V fragments are independent of the
    // softmax, so their latency hides under score WMMAs + LDS + expf work.
    v16h bk00 = load_b32x16(kT, HD);            // keys 0-15, d 0-31
    v16h bk01 = load_b32x16(kT + 32, HD);       // keys 0-15, d 32-63
    v16h bk10 = load_b32x16(kT + 16 * HD, HD);  // keys 16-31
    v16h bk11 = load_b32x16(kT + 16 * HD + 32, HD);
    v16h bv0 = load_b32x16(vT + (size_t)0 * 16 * vld, vld);
    v16h bv1 = load_b32x16(vT + (size_t)1 * 16 * vld, vld);
    v16h bv2 = load_b32x16(vT + (size_t)2 * 16 * vld, vld);
    v16h bv3 = load_b32x16(vT + (size_t)3 * 16 * vld, vld);

    v8f t0 = {}, t1 = {};
    t0 = wmma_f16(aq0, bk00, t0);
    t1 = wmma_f16(aq0, bk10, t1);
    t0 = wmma_f16(aq1, bk01, t0);
    t1 = wmma_f16(aq1, bk11, t1);
#pragma unroll
    for (int r = 0; r < 8; ++r) {             // D layout: lane=N, VGPR=M
      sc[rb + r][n]      = t0[r];
      sc[rb + r][16 + n] = t1[r];
    }
    asm volatile("s_wait_dscnt 0" ::: "memory");

    if (lane < 16) {                          // row-wise masked online softmax
      const int i = q0 + lane;
      v4f sv[8];                              // vectorized LDS reads (b128)
#pragma unroll
      for (int jv = 0; jv < 8; ++jv) sv[jv] = *(const v4f*)&sc[lane][jv * 4];
      float vals[32];
      float mx = -3.0e38f;
#pragma unroll
      for (int j = 0; j < 32; ++j) {
        const int jj = jbase + j * jstep;
        const float s = sv[j >> 2][j & 3] * 0.125f; // 1/sqrt(64)
        const int band = (jj - i <= WIN) && (i - jj <= WIN);
        int valid;
        if (mode == 0)      valid = band | ((i & 63) == 0) | ((jj & 63) == 0);
        else if (mode == 1) valid = band | ((jj & 63) == 0);
        else                valid = !((jj >= klo) && (jj < khi)); // compact: skip visited
        vals[j] = valid ? s : NEG_INF;
        mx = fmaxf(mx, vals[j]);
      }
      const float m_new = fmaxf(m_i, mx);
      const float alpha = __expf(m_i - m_new);
      float lsum = 0.0f;
#pragma unroll
      for (int jv = 0; jv < 4; ++jv) {        // vectorized LDS writes (b128)
        v8h pk;
#pragma unroll
        for (int e = 0; e < 8; ++e) {
          const float p = __expf(vals[jv * 8 + e] - m_new);
          lsum += p;
          pk[e] = (half_t)p;
        }
        *(v8h*)&pf[lane][jv * 8] = pk;
      }
      l_i = l_i * alpha + lsum;
      m_i = m_new;
      alf[lane] = alpha;
    }
    asm volatile("s_wait_dscnt 0" ::: "memory");

    float al[8];
#pragma unroll
    for (int r = 0; r < 8; ++r) al[r] = alf[rb + r];
#pragma unroll
    for (int t = 0; t < 4; ++t)
#pragma unroll
      for (int r = 0; r < 8; ++r) acc[t][r] *= al[r];

    v16h pA;                                  // P as A-frag (2x b128 LDS loads)
    {
      const int m = lane & 15, ko = (lane >> 4) * 8;
      v8h plo = *(const v8h*)&pf[m][ko];
      v8h phi = *(const v8h*)&pf[m][16 + ko];
#pragma unroll
      for (int i2 = 0; i2 < 8; ++i2) { pA[i2] = plo[i2]; pA[8 + i2] = phi[i2]; }
    }
    acc[0] = wmma_f16(pA, bv0, acc[0]);
    acc[1] = wmma_f16(pA, bv1, acc[1]);
    acc[2] = wmma_f16(pA, bv2, acc[2]);
    acc[3] = wmma_f16(pA, bv3, acc[3]);
  };

  const half_t* KhH = Kh + (size_t)h * S_LEN * HD;
  const half_t* VtH = Vt + (size_t)h * HD * S_LEN;

  if ((q0 & 63) == 0) {
    // tile contains a global row -> masked dense sweep (exact)
    for (int kb = 0; kb < S_LEN / 32; ++kb) {
      if (kb + 1 < S_LEN / 32)
        __builtin_prefetch(KhH + (size_t)(kb + 1) * 32 * HD, 0, 1);
      process(KhH + (size_t)kb * 32 * HD, VtH + kb * 32, S_LEN,
              kb * 32, 1, /*mode=*/0, 0, 0);
    }
  } else {
    int lo = q0 - WIN;       if (lo < 0) lo = 0;
    int hi = q0 + 15 + WIN;  if (hi > S_LEN - 1) hi = S_LEN - 1;
    const int blo = lo >> 5, bhi = hi >> 5;
    for (int kb = blo; kb <= bhi; ++kb) {     // local band (+global cols inside)
      if (kb < bhi)
        __builtin_prefetch(KhH + (size_t)(kb + 1) * 32 * HD, 0, 1);
      process(KhH + (size_t)kb * 32 * HD, VtH + kb * 32, S_LEN,
              kb * 32, 1, /*mode=*/1, 0, 0);
    }
    // compact block of all 32 global keys, excluding already-visited range
    process(Kg + (size_t)h * NGLOB * HD,
            Vgt + (size_t)h * HD * NGLOB, NGLOB,
            0, GSTRIDE, /*mode=*/2, blo * 32, (bhi + 1) * 32);
  }

  if (lane < 16) linv[lane] = 1.0f / l_i;
  asm volatile("s_wait_dscnt 0" ::: "memory");
  float li[8];
#pragma unroll
  for (int r = 0; r < 8; ++r) li[r] = linv[rb + r];
#pragma unroll
  for (int t = 0; t < 4; ++t)
#pragma unroll
    for (int r = 0; r < 8; ++r)
      attn_h[(size_t)(q0 + rb + r) * HID + h * HD + t * 16 + n] =
          (half_t)(acc[t][r] * li[r]);
}

// out = attn @ Wout^T (f32 output)
__global__ __launch_bounds__(128) void out_gemm(
    const half_t* __restrict__ ah, const half_t* __restrict__ wo,
    float* __restrict__ out) {
  const int wave = blockIdx.x * (blockDim.x >> 5) + (threadIdx.x >> 5);
  const int NT = HID / 64;                    // 16 col-tiles
  const int mt = wave / NT, nt = wave % NT;
  const int s0 = mt * 32, o0 = nt * 64;
  const int lane = threadIdx.x & 31;
  const int n = lane & 15;
  const int rb = (lane >> 4) * 8;

  v8f acc[2][4] = {};
  gemm_32x64_acc(ah + (size_t)s0 * HID, wo + (size_t)o0 * HID, acc);

#pragma unroll
  for (int i = 0; i < 2; ++i)
#pragma unroll
    for (int t = 0; t < 4; ++t)
#pragma unroll
      for (int r = 0; r < 8; ++r)
        out[(size_t)(s0 + i * 16 + rb + r) * HID + o0 + t * 16 + n] =
            acc[i][t][r];
}

// ---------------------------------------------------------------- launcher

extern "C" void kernel_launch(void* const* d_in, const int* in_sizes, int n_in,
                              void* d_out, int out_size, void* d_ws, size_t ws_size,
                              hipStream_t stream) {
  const float* x    = (const float*)d_in[0];   // [1,2048,1024]
  const float* Wqkv = (const float*)d_in[1];   // [3072,1024]
  const float* Wout = (const float*)d_in[2];   // [1024,1024]
  float* out = (float*)d_out;                  // [1,2048,1024]

  char* p = (char*)d_ws;
  half_t* xh    = (half_t*)p; p += (size_t)S_LEN * HID * 2;
  half_t* wqkvh = (half_t*)p; p += (size_t)3 * HID * HID * 2;
  half_t* wouth = (half_t*)p; p += (size_t)HID * HID * 2;
  half_t* Qh    = (half_t*)p; p += (size_t)NH * S_LEN * HD * 2;
  half_t* Kh    = (half_t*)p; p += (size_t)NH * S_LEN * HD * 2;
  half_t* Vt    = (half_t*)p; p += (size_t)NH * S_LEN * HD * 2;
  half_t* Kg    = (half_t*)p; p += (size_t)NH * NGLOB * HD * 2;
  half_t* Vgt   = (half_t*)p; p += (size_t)NH * NGLOB * HD * 2;
  half_t* attn  = (half_t*)p; p += (size_t)S_LEN * HID * 2;

  const int nx = S_LEN * HID, nq = 3 * HID * HID, no = HID * HID;
  cvt_f32_f16<<<nx / 4 / 256, 256, 0, stream>>>(x, xh, nx);
  cvt_f32_f16<<<nq / 4 / 256, 256, 0, stream>>>(Wqkv, wqkvh, nq);
  cvt_f32_f16<<<no / 4 / 256, 256, 0, stream>>>(Wout, wouth, no);

  qkv_gemm<<<((S_LEN / 32) * 48) / 4, 128, 0, stream>>>(xh, wqkvh, Qh, Kh, Vt);
  gather_glb<<<(NH * NGLOB * HD + 255) / 256, 256, 0, stream>>>(Kh, Vt, Kg, Vgt);
  attn_sparse<<<NH * (S_LEN / 16), 32, 0, stream>>>(Qh, Kh, Vt, Kg, Vgt, attn);
  out_gemm<<<((S_LEN / 32) * 16) / 4, 128, 0, stream>>>(attn, wouth, out);
}
